// TGS_4166118277863
// MI455X (gfx1250) — matrix operand order repaired and checked by reading, not compiled, not run on hardware
//
#include <hip/hip_runtime.h>
#include <hip/hip_bf16.h>

typedef __attribute__((ext_vector_type(16))) _Float16 v16h;
typedef __attribute__((ext_vector_type(8)))  _Float16 v8h;
typedef __attribute__((ext_vector_type(4)))  _Float16 v4h;
typedef __attribute__((ext_vector_type(8)))  float    v8f;
typedef __attribute__((ext_vector_type(4)))  float    v4f;

#define Dd   128
#define Ee   20
#define Kn   10
#define C1   276      // 2D + E
#define C1P  288      // padded to 9 * 32
#define C2   384      // 3D = 12 * 32
#define KT1  9        // C1P / 32
#define KT2  12       // C2 / 32

// packed weight sizes (halves), per layer
#define PW1_LAYER (KT1 * 8 * 32 * 16)   // 36864
#define PW2_LAYER (KT2 * 8 * 32 * 16)   // 49152

__device__ __forceinline__ v8f wmma_f16(v16h a, v16h b, v8f c) {
  return __builtin_amdgcn_wmma_f32_16x16x32_f16(
      /*neg_a=*/false, a, /*neg_b=*/false, b,
      /*c_mod=*/(short)0, c, /*reuse_a=*/false, /*reuse_b=*/false);
}

__device__ __forceinline__ v16h ld_a_frag(const _Float16* p0, const _Float16* p1) {
  // Two contiguous 8-half (16B) chunks -> 16-half A fragment
  v8h lo = *(const v8h*)p0;
  v8h hi = *(const v8h*)p1;
  v16h r;
#pragma unroll
  for (int i = 0; i < 8; ++i) { r[i] = lo[i]; r[8 + i] = hi[i]; }
  return r;
}

__device__ __forceinline__ v4h cvt4(v4f a) {
  v4h h;
#pragma unroll
  for (int i = 0; i < 4; ++i) h[i] = (_Float16)a[i];
  return h;
}

// -------- weight packing into per-lane B-fragment order --------------------
// B fragment (16-bit, 32x16 K x N): lane L holds, for i in [0,16):
//   b[i] = W[kt*32 + (L/16)*16 + i][nt*16 + (L%16)]
// Packed layout: ((((layer*KT + kt)*8 + nt)*32 + lane)*16 + i)
__global__ void pack_weights_kernel(const float* __restrict__ lin1_w,
                                    const float* __restrict__ lin2_w,
                                    _Float16* __restrict__ pw1,
                                    _Float16* __restrict__ pw2) {
  int idx = blockIdx.x * blockDim.x + threadIdx.x;
  const int N1 = 2 * PW1_LAYER;
  const int N2 = 2 * PW2_LAYER;
  if (idx < N1) {
    int h = idx & 15, lane = (idx >> 4) & 31, nt = (idx >> 9) & 7;
    int t = idx >> 12;                 // layer*KT1 + kt
    int kt = t % KT1, l = t / KT1;
    int Kk = kt * 32 + (lane >> 4) * 16 + h;
    int n  = nt * 16 + (lane & 15);
    float v = (Kk < C1) ? lin1_w[(l * C1 + Kk) * Dd + n] : 0.0f;
    pw1[idx] = (_Float16)v;
  } else if (idx < N1 + N2) {
    int j = idx - N1;
    int h = j & 15, lane = (j >> 4) & 31, nt = (j >> 9) & 7;
    int t = j >> 12;
    int kt = t % KT2, l = t / KT2;
    int Kk = kt * 32 + (lane >> 4) * 16 + h;
    int n  = nt * 16 + (lane & 15);
    pw2[j] = (_Float16)lin2_w[(l * C2 + Kk) * Dd + n];
  }
}

// -------- one aggregation level: 16 rows per block, 128 threads (4 waves) --
__global__ __launch_bounds__(128)
void tgs_level_kernel(const float* __restrict__ nbr_src,       // node_features (L1) or emb1 (L2)
                      const float* __restrict__ node_features, // center features
                      const float* __restrict__ edge_features,
                      const int*   __restrict__ center_ids,    // neighbors1 flat (L1) / source_nodes (L2)
                      const int*   __restrict__ nbr_ids,       // neighbors2 flat (L1) / unused (L2)
                      const int*   __restrict__ edge_idxs,
                      const float* __restrict__ edge_times,
                      const float* __restrict__ timestamps,
                      const float* __restrict__ time_w,
                      const float* __restrict__ time_b,
                      const _Float16* __restrict__ pw1,        // this layer's packed W1
                      const _Float16* __restrict__ pw2,        // this layer's packed W2
                      const float* __restrict__ bias1,         // this layer's lin1_b [128]
                      const float* __restrict__ bias2,         // this layer's lin2_b [128]
                      float* __restrict__ out,
                      int level) {
  __shared__ alignas(16) _Float16 sA[16][C1P]; // summed concat vectors (f16)
  __shared__ alignas(16) _Float16 sB[16][C2];  // [src_feat | cos(b) | relu-sum]
  __shared__ int   sNbr[16][Kn];
  __shared__ int   sEidx[16][Kn];
  __shared__ float sEt[16][Kn];
  __shared__ float sT[16];
  __shared__ int   sCenter[16];

  const int tid = threadIdx.x;
  const int r0  = blockIdx.x * 16;

  // ---- stage per-row index data --------------------------------------
  for (int i = tid; i < 16 * Kn; i += 128) {
    int row = i / Kn, k = i % Kn;
    int r = r0 + row;
    sNbr[row][k]  = (level == 1) ? nbr_ids[r * Kn + k] : (r * Kn + k);
    sEidx[row][k] = edge_idxs[r * Kn + k];
    sEt[row][k]   = edge_times[r * Kn + k];
  }
  if (tid < 16) {
    int r = r0 + tid;
    sT[tid]      = (level == 1) ? timestamps[r / Kn] : timestamps[r];
    sCenter[tid] = center_ids[r];
  }
  __syncthreads();

  // ---- phase 1: gather + K-sum (linear-layer fold), vectorized b128 ----
  {
    const int row = tid >> 3;      // 16 rows
    const int c   = tid & 7;       // 8 threads per row
    const float tr    = sT[row];
    const int  center = sCenter[row];

    int   nb[Kn], ei[Kn];
    float et[Kn];
#pragma unroll
    for (int k = 0; k < Kn; ++k) {
      nb[k] = sNbr[row][k];
      ei[k] = sEidx[row][k];
      et[k] = sEt[row][k];
    }

    // sB cols [0,256): center features + time_enc(0) = cos(time_b)
#pragma unroll
    for (int i = 0; i < 4; ++i) {               // 4 float4 chunks per thread
      int j = c * 4 + 32 * i;                   // covers 0..127
      v4f f = *(const v4f*)(node_features + center * Dd + j);
      *(v4h*)&sB[row][j] = cvt4(f);
      v4f b4 = *(const v4f*)(time_b + j);
      v4f cb;
#pragma unroll
      for (int q = 0; q < 4; ++q) cb[q] = __cosf(b4[q]);
      *(v4h*)&sB[row][128 + j] = cvt4(cb);
    }

    // sA cols [0,128): sum_k nbr feature rows (b128 gathers)
#pragma unroll
    for (int i = 0; i < 4; ++i) {
      int j = c * 4 + 32 * i;
      v4f acc = {0.0f, 0.0f, 0.0f, 0.0f};
#pragma unroll
      for (int k = 0; k < Kn; ++k)
        acc += *(const v4f*)(nbr_src + nb[k] * Dd + j);
      *(v4h*)&sA[row][j] = cvt4(acc);
    }

    // sA cols [128,256): sum_k cos((tr - et_k)*w + b)
#pragma unroll
    for (int i = 0; i < 4; ++i) {
      int jj = c * 4 + 32 * i;
      v4f w4 = *(const v4f*)(time_w + jj);
      v4f b4 = *(const v4f*)(time_b + jj);
      v4f acc = {0.0f, 0.0f, 0.0f, 0.0f};
#pragma unroll
      for (int k = 0; k < Kn; ++k) {
        float dt = tr - et[k];
#pragma unroll
        for (int q = 0; q < 4; ++q) acc[q] += __cosf(dt * w4[q] + b4[q]);
      }
      *(v4h*)&sA[row][128 + jj] = cvt4(acc);
    }

    // sA cols [256,276): sum_k edge features (5 chunks), [276,288): zeros
    {
      int j = 256 + c * 4;
      if (c < 5) {
        int jj = c * 4;
        v4f acc = {0.0f, 0.0f, 0.0f, 0.0f};
#pragma unroll
        for (int k = 0; k < Kn; ++k)
          acc += *(const v4f*)(edge_features + ei[k] * Ee + jj);
        *(v4h*)&sA[row][j] = cvt4(acc);
      } else {
        v4h z = {};
        *(v4h*)&sA[row][j] = z;
      }
    }
  }
  __syncthreads();

  const int lane = tid & 31;
  const int wave = tid >> 5;
  const int m    = lane & 15;
  const int hig  = lane >> 4;          // lane half select
  const int nt0  = 2 * wave;           // each wave owns 2 N-tiles

  // ---- GEMM1: (16 x 288) @ (288 x 128), + 10*b1, ReLU -> sB[.,256:384)
  {
    v8f c0 = {}, c1 = {};
    for (int kt = 0; kt < KT1; ++kt) {
      int colA = kt * 32 + hig * 16;
      v16h a = ld_a_frag(&sA[m][colA], &sA[m][colA + 8]);
      const _Float16* bp = pw1 + ((kt * 8 + nt0) * 32 + lane) * 16;
      v16h b0 = *(const v16h*)bp;
      v16h b1 = *(const v16h*)(bp + 32 * 16);
      c0 = wmma_f16(a, b0, c0);
      c1 = wmma_f16(a, b1, c1);
    }
    int n0 = nt0 * 16 + (lane & 15);
    int n1 = n0 + 16;
    float bb0 = 10.0f * bias1[n0], bb1 = 10.0f * bias1[n1];
#pragma unroll
    for (int v = 0; v < 8; ++v) {
      int mrow = v + 8 * hig;
      float x0 = c0[v] + bb0; x0 = x0 > 0.0f ? x0 : 0.0f;
      float x1 = c1[v] + bb1; x1 = x1 > 0.0f ? x1 : 0.0f;
      sB[mrow][256 + n0] = (_Float16)x0;
      sB[mrow][256 + n1] = (_Float16)x1;
    }
  }
  __syncthreads();

  // ---- GEMM2: (16 x 384) @ (384 x 128), + b2 -> out -------------------
  {
    v8f c0 = {}, c1 = {};
    for (int kt = 0; kt < KT2; ++kt) {
      int colA = kt * 32 + hig * 16;
      v16h a = ld_a_frag(&sB[m][colA], &sB[m][colA + 8]);
      const _Float16* bp = pw2 + ((kt * 8 + nt0) * 32 + lane) * 16;
      v16h b0 = *(const v16h*)bp;
      v16h b1 = *(const v16h*)(bp + 32 * 16);
      c0 = wmma_f16(a, b0, c0);
      c1 = wmma_f16(a, b1, c1);
    }
    int n0 = nt0 * 16 + (lane & 15);
    int n1 = n0 + 16;
    float bb0 = bias2[n0], bb1 = bias2[n1];
#pragma unroll
    for (int v = 0; v < 8; ++v) {
      int mrow = v + 8 * hig;
      out[(r0 + mrow) * Dd + n0] = c0[v] + bb0;
      out[(r0 + mrow) * Dd + n1] = c1[v] + bb1;
    }
  }
}

extern "C" void kernel_launch(void* const* d_in, const int* in_sizes, int n_in,
                              void* d_out, int out_size, void* d_ws, size_t ws_size,
                              hipStream_t stream) {
  const float* node_features = (const float*)d_in[0];
  const float* edge_features = (const float*)d_in[1];
  const int*   source_nodes  = (const int*)d_in[2];
  const float* timestamps    = (const float*)d_in[3];
  const int*   neighbors1    = (const int*)d_in[4];
  const int*   edge_idxs1    = (const int*)d_in[5];
  const float* edge_times1   = (const float*)d_in[6];
  const int*   neighbors2    = (const int*)d_in[7];
  const int*   edge_idxs2    = (const int*)d_in[8];
  const float* edge_times2   = (const float*)d_in[9];
  const float* time_w        = (const float*)d_in[10];
  const float* time_b        = (const float*)d_in[11];
  const float* lin1_w        = (const float*)d_in[12];
  const float* lin1_b        = (const float*)d_in[13];
  const float* lin2_w        = (const float*)d_in[14];
  const float* lin2_b        = (const float*)d_in[15];

  const int N  = 4096;
  const int NK = N * Kn;                         // 40960 first-hop rows

  // workspace layout (all offsets 32B-aligned):
  //   emb1: NK * 128 f32  (20,971,520 B)
  //   pw1 : 2 * PW1_LAYER f16
  //   pw2 : 2 * PW2_LAYER f16
  char* ws = (char*)d_ws;
  float*    emb1 = (float*)ws;
  _Float16* pw1  = (_Float16*)(ws + (size_t)NK * Dd * sizeof(float));
  _Float16* pw2  = (_Float16*)((char*)pw1 + (size_t)2 * PW1_LAYER * sizeof(_Float16));

  // 1) pack weights into WMMA B-fragment order (f16)
  {
    int total = 2 * PW1_LAYER + 2 * PW2_LAYER;   // 172032
    pack_weights_kernel<<<(total + 255) / 256, 256, 0, stream>>>(lin1_w, lin2_w, pw1, pw2);
  }

  // 2) level 1: embed the N*K first-hop neighbors (layer index 0) -> emb1
  tgs_level_kernel<<<NK / 16, 128, 0, stream>>>(
      node_features, node_features, edge_features,
      neighbors1, neighbors2, edge_idxs2, edge_times2, timestamps,
      time_w, time_b,
      pw1, pw2, lin1_b, lin2_b,
      emb1, /*level=*/1);

  // 3) level 2: embed the N source nodes (layer index 1) -> d_out
  tgs_level_kernel<<<N / 16, 128, 0, stream>>>(
      emb1, node_features, edge_features,
      source_nodes, nullptr, edge_idxs1, edge_times1, timestamps,
      time_w, time_b,
      pw1 + PW1_LAYER, pw2 + PW2_LAYER, lin1_b + Dd, lin2_b + Dd,
      (float*)d_out, /*level=*/2);
}
